// T2SModel_19181323944342
// MI455X (gfx1250) — compile-verified, matrix operand order checked by reading, not run on hardware
//
#include <hip/hip_runtime.h>
#include <hip/hip_bf16.h>
#include <math.h>

// ---------------- model constants ----------------
#define D      512
#define HEADS  16
#define DH     32
#define FF     2048
#define LAYERS 24
#define VOC    1025
#define XLEN   200
#define PLEN   150
#define SEQ    350          // XLEN + PLEN
#define SEQP   384          // SEQ padded to multiple of 64 for guard-free GEMM
#define TMAX   381          // SEQ + NEWTOK - 1
#define NEWTOK 32
#define BERTP  256          // 200 bert rows padded

typedef __bf16 bf16_t;
typedef bf16_t v16bf  __attribute__((ext_vector_type(16)));
typedef bf16_t bf16x8 __attribute__((ext_vector_type(8)));
typedef bf16_t bf16x4 __attribute__((ext_vector_type(4)));
typedef float  v8f    __attribute__((ext_vector_type(8)));

// ---------------- helpers ----------------
__device__ __forceinline__ float pe_val(int pos, int d) {
    int i2 = (d >> 1) << 1;
    float div = expf(-(float)i2 * (logf(10000.0f) / (float)D));
    float a = (float)pos * div;
    return (d & 1) ? cosf(a) : sinf(a);
}

__device__ __forceinline__ unsigned hash_u32(unsigned a) {
    a ^= a >> 16; a *= 0x7feb352du;
    a ^= a >> 15; a *= 0x846ca68bu;
    a ^= a >> 16;
    return a;
}

// ---------------- WMMA GEMM: C[M,N] = A[M,K] * B[K,N] (+bias, relu) ----------------
// Requirements: M % 64 == 0, N % 128 == 0, K % 32 == 0 (caller pads buffers).
// Block = 128 threads (4 waves). Block tile 64x128, wave tile 32x64
// = 2x4 WMMA 16x16x32 bf16 accumulators -> 8 v_wmma per k-step.
#define BM 64
#define BN 128
#define BK 32

__device__ __forceinline__ v16bf frag_ld(const bf16_t* p) {
    // two 16B LDS vector loads, joined register-to-register
    bf16x8 lo = *(const bf16x8*)p;
    bf16x8 hi = *(const bf16x8*)(p + 16);
    return __builtin_shufflevector(lo, hi, 0, 1, 2, 3, 4, 5, 6, 7,
                                            8, 9, 10, 11, 12, 13, 14, 15);
}

__global__ void __launch_bounds__(128, 1)
gemm_wmma(const float* __restrict__ A, const float* __restrict__ B,
          const float* __restrict__ bias, float* __restrict__ C,
          int M, int N, int K, int relu)
{
    __shared__ bf16_t As[BM][BK + 8];   // [row][k]
    __shared__ bf16_t Bs[BN][BK + 8];   // [col][k]  (B transposed in LDS)

    const int tid    = threadIdx.x;
    const int lane   = tid & 31;
    const int wave   = tid >> 5;
    const int blockM = blockIdx.y * BM;
    const int blockN = blockIdx.x * BN;
    const int waveM  = (wave >> 1) * 32;
    const int waveN  = (wave & 1) * 64;
    const int lmod   = lane & 15;
    const int lhalf  = lane >> 4;
    const int kb     = lhalf * 8;

    v8f zero = {};
    v8f acc[2][4];
    for (int i = 0; i < 2; ++i)
        for (int j = 0; j < 4; ++j) acc[i][j] = zero;

    for (int k0 = 0; k0 < K; k0 += BK) {
        if (k0 + BK < K) {
            // gfx1250 global_prefetch_b8 of next tiles (WGP scope -> all levels)
            __builtin_prefetch(&A[(size_t)(blockM + (tid >> 1)) * K + k0 + BK], 0, 3);
            __builtin_prefetch(&B[(size_t)(k0 + BK + (tid >> 2)) * N + blockN + (tid & 3) * 32], 0, 3);
        }
        // A tile 64x32: 512 float4 chunks, 4 per thread, guard-free
#pragma unroll
        for (int i = 0; i < 4; ++i) {
            int chunk = tid + i * 128;
            int r = chunk >> 3;
            int c = (chunk & 7) * 4;
            float4 v = *(const float4*)&A[(size_t)(blockM + r) * K + k0 + c];
            bf16x4 p;
            p[0] = (bf16_t)v.x; p[1] = (bf16_t)v.y;
            p[2] = (bf16_t)v.z; p[3] = (bf16_t)v.w;
            *(bf16x4*)&As[r][c] = p;          // ds_store_b64
        }
        // B tile 32x128: each thread loads a 4-deep k-column (4 coalesced b32
        // loads across lanes), packs to bf16x4, single ds_store_b64 per chunk
#pragma unroll
        for (int i = 0; i < 8; ++i) {
            int chunk = tid + i * 128;
            int n  = chunk & 127;
            int kg = (chunk >> 7) * 4;        // 0,4,...,28
            const float* bp = &B[(size_t)(k0 + kg) * N + blockN + n];
            bf16x4 p;
            p[0] = (bf16_t)bp[0];
            p[1] = (bf16_t)bp[(size_t)N];
            p[2] = (bf16_t)bp[(size_t)2 * N];
            p[3] = (bf16_t)bp[(size_t)3 * N];
            *(bf16x4*)&Bs[n][kg] = p;         // ds_store_b64
        }
        __syncthreads();

        // fragments per CDNA5 16-bit layout: lane l -> row/col l%16,
        // k in {kb..kb+7} U {kb+16..kb+23}, kb = (l/16)*8
        v16bf a0 = frag_ld(&As[waveM + lmod][kb]);
        v16bf a1 = frag_ld(&As[waveM + 16 + lmod][kb]);
        v16bf b0 = frag_ld(&Bs[waveN + lmod][kb]);
        v16bf b1 = frag_ld(&Bs[waveN + 16 + lmod][kb]);
        v16bf b2 = frag_ld(&Bs[waveN + 32 + lmod][kb]);
        v16bf b3 = frag_ld(&Bs[waveN + 48 + lmod][kb]);

        acc[0][0] = __builtin_amdgcn_wmma_f32_16x16x32_bf16(false, a0, false, b0, (short)0, acc[0][0], false, false);
        acc[0][1] = __builtin_amdgcn_wmma_f32_16x16x32_bf16(false, a0, false, b1, (short)0, acc[0][1], false, false);
        acc[0][2] = __builtin_amdgcn_wmma_f32_16x16x32_bf16(false, a0, false, b2, (short)0, acc[0][2], false, false);
        acc[0][3] = __builtin_amdgcn_wmma_f32_16x16x32_bf16(false, a0, false, b3, (short)0, acc[0][3], false, false);
        acc[1][0] = __builtin_amdgcn_wmma_f32_16x16x32_bf16(false, a1, false, b0, (short)0, acc[1][0], false, false);
        acc[1][1] = __builtin_amdgcn_wmma_f32_16x16x32_bf16(false, a1, false, b1, (short)0, acc[1][1], false, false);
        acc[1][2] = __builtin_amdgcn_wmma_f32_16x16x32_bf16(false, a1, false, b2, (short)0, acc[1][2], false, false);
        acc[1][3] = __builtin_amdgcn_wmma_f32_16x16x32_bf16(false, a1, false, b3, (short)0, acc[1][3], false, false);
        __syncthreads();
    }

    // epilogue: lane l -> n = l%16; vgpr j -> m = j + 8*(l/16); bias hoisted per nt
#pragma unroll
    for (int nt = 0; nt < 4; ++nt) {
        int gn = blockN + waveN + nt * 16 + lmod;
        float badd = bias ? bias[gn] : 0.0f;
#pragma unroll
        for (int mt = 0; mt < 2; ++mt)
#pragma unroll
            for (int j = 0; j < 8; ++j) {
                int gm = blockM + waveM + mt * 16 + lhalf * 8 + j;
                float v = acc[mt][nt][j] + badd;
                if (relu) v = fmaxf(v, 0.0f);
                C[(size_t)gm * N + gn] = v;
            }
    }
}

// ---------------- f32 -> bf16 weight conversion ----------------
__global__ void __launch_bounds__(256)
cvt_bf16(const float* __restrict__ src, bf16_t* __restrict__ dst, int n4)
{
    int i = blockIdx.x * 256 + threadIdx.x;   // one float4 per thread
    if (i < n4) {
        float4 v = *(const float4*)&src[(size_t)i * 4];
        bf16x4 p;
        p[0] = (bf16_t)v.x; p[1] = (bf16_t)v.y;
        p[2] = (bf16_t)v.z; p[3] = (bf16_t)v.w;
        *(bf16x4*)&dst[(size_t)i * 4] = p;
    }
}

// ---------------- bert rows -> padded buffer ----------------
__global__ void __launch_bounds__(256)
pack_bert(const float* __restrict__ ref_bert, const float* __restrict__ text_bert,
          float* __restrict__ dst)
{
    int r = blockIdx.x, tid = threadIdx.x;    // 200 rows of 1024
    const float* src = (r < 80) ? (ref_bert + (size_t)r * 1024)
                                : (text_bert + (size_t)(r - 80) * 1024);
    for (int i = tid; i < 1024; i += 256) dst[(size_t)r * 1024 + i] = src[i];
}

// ---------------- embedding / setup ----------------
__global__ void __launch_bounds__(512)
embed_kernel(const float* __restrict__ te, const float* __restrict__ ae,
             const float* __restrict__ bproj, const float* __restrict__ bb,
             const float* __restrict__ alpha_t, const float* __restrict__ alpha_a,
             const int* __restrict__ prompts, const int* __restrict__ ref_seq,
             const int* __restrict__ text_seq, float* __restrict__ h)
{
    int r = blockIdx.x, d = threadIdx.x;
    if (r < XLEN) {
        int tok = (r < 80) ? ref_seq[r] : text_seq[r - 80];
        h[(size_t)r * D + d] = te[(size_t)tok * D + d] + bproj[(size_t)r * D + d]
                             + bb[d] + alpha_t[0] * pe_val(r, d);
    } else {
        int p = r - XLEN;
        int tok = prompts[p];
        h[(size_t)r * D + d] = ae[(size_t)tok * D + d] + alpha_a[0] * pe_val(p, d);
    }
}

__global__ void init_ybuf(const int* __restrict__ prompts, int* __restrict__ ybuf) {
    int i = threadIdx.x;
    if (i < PLEN + NEWTOK) ybuf[i] = (i < PLEN) ? prompts[i] : 0;
}

// ---------------- prefill attention (also fills KV cache) ----------------
__global__ void __launch_bounds__(32)
attn_prefill(const float* __restrict__ qkv, float* __restrict__ kc,
             float* __restrict__ vc, float* __restrict__ attout)
{
    int row = blockIdx.x, head = blockIdx.y, lane = threadIdx.x;
    __shared__ float qs[DH];
    __shared__ float sm[32], ss[32];
    __shared__ float sacc[32][DH + 1];

    qs[lane] = qkv[(size_t)row * 3 * D + head * DH + lane];
    kc[((size_t)row * HEADS + head) * DH + lane] = qkv[(size_t)row * 3 * D + D     + head * DH + lane];
    vc[((size_t)row * HEADS + head) * DH + lane] = qkv[(size_t)row * 3 * D + 2 * D + head * DH + lane];
    __syncthreads();

    int limit = (row < XLEN) ? XLEN : (row + 1);
    const float scale = 0.1767766953f;   // 1/sqrt(32)
    float m = -1e30f, s = 0.0f, acc[DH];
    for (int d = 0; d < DH; ++d) acc[d] = 0.0f;

    for (int t = lane; t < limit; t += 32) {
        const float* kr = qkv + (size_t)t * 3 * D + D     + head * DH;
        const float* vr = qkv + (size_t)t * 3 * D + 2 * D + head * DH;
        float dot = 0.0f;
        for (int d = 0; d < DH; ++d) dot += qs[d] * kr[d];
        dot *= scale;
        float nm = fmaxf(m, dot);
        float fs = expf(m - nm);
        float w  = expf(dot - nm);
        s = s * fs + w;
        for (int d = 0; d < DH; ++d) acc[d] = acc[d] * fs + w * vr[d];
        m = nm;
    }
    sm[lane] = m; ss[lane] = s;
    for (int d = 0; d < DH; ++d) sacc[lane][d] = acc[d];
    __syncthreads();

    float gm = -1e30f;
    for (int j = 0; j < 32; ++j) gm = fmaxf(gm, sm[j]);
    float tot = 0.0f;
    for (int j = 0; j < 32; ++j) tot += ss[j] * expf(sm[j] - gm);
    float o = 0.0f;
    for (int j = 0; j < 32; ++j) o += expf(sm[j] - gm) * sacc[j][lane];
    attout[(size_t)row * D + head * DH + lane] = o / tot;
}

// ---------------- decode attention (single query, KV append) ----------------
__global__ void __launch_bounds__(32)
attn_decode(const float* __restrict__ qkvvec, float* __restrict__ kc,
            float* __restrict__ vc, float* __restrict__ ovec, int pos)
{
    int head = blockIdx.x, lane = threadIdx.x;
    __shared__ float qs[DH], kcur[DH], vcur[DH];
    __shared__ float sm[32], ss[32];
    __shared__ float sacc[32][DH + 1];

    qs[lane]   = qkvvec[head * DH + lane];
    kcur[lane] = qkvvec[D     + head * DH + lane];
    vcur[lane] = qkvvec[2 * D + head * DH + lane];
    kc[((size_t)pos * HEADS + head) * DH + lane] = kcur[lane];
    vc[((size_t)pos * HEADS + head) * DH + lane] = vcur[lane];
    __syncthreads();

    const float scale = 0.1767766953f;
    float m = -1e30f, s = 0.0f, acc[DH];
    for (int d = 0; d < DH; ++d) acc[d] = 0.0f;

    for (int t = lane; t <= pos; t += 32) {
        const float* kr = (t == pos) ? kcur : (kc + ((size_t)t * HEADS + head) * DH);
        const float* vr = (t == pos) ? vcur : (vc + ((size_t)t * HEADS + head) * DH);
        float dot = 0.0f;
        for (int d = 0; d < DH; ++d) dot += qs[d] * kr[d];
        dot *= scale;
        float nm = fmaxf(m, dot);
        float fs = expf(m - nm);
        float w  = expf(dot - nm);
        s = s * fs + w;
        for (int d = 0; d < DH; ++d) acc[d] = acc[d] * fs + w * vr[d];
        m = nm;
    }
    sm[lane] = m; ss[lane] = s;
    for (int d = 0; d < DH; ++d) sacc[lane][d] = acc[d];
    __syncthreads();

    float gm = -1e30f;
    for (int j = 0; j < 32; ++j) gm = fmaxf(gm, sm[j]);
    float tot = 0.0f;
    for (int j = 0; j < 32; ++j) tot += ss[j] * expf(sm[j] - gm);
    float o = 0.0f;
    for (int j = 0; j < 32; ++j) o += expf(sm[j] - gm) * sacc[j][lane];
    ovec[head * DH + lane] = o / tot;
}

// ---------------- residual + post-LN (grid = rows, block = 128) ----------------
__global__ void __launch_bounds__(128)
add_ln(const float* __restrict__ base, const float* __restrict__ add,
       const float* __restrict__ g, const float* __restrict__ b,
       float* __restrict__ out)
{
    int row = blockIdx.x, tid = threadIdx.x;
    const float* br = base + (size_t)row * D;
    const float* ar = add  + (size_t)row * D;
    float v[4];
    for (int i = 0; i < 4; ++i) v[i] = br[tid * 4 + i] + ar[tid * 4 + i];

    __shared__ float red[128];
    float s = v[0] + v[1] + v[2] + v[3];
    red[tid] = s; __syncthreads();
    for (int o = 64; o > 0; o >>= 1) { if (tid < o) red[tid] += red[tid + o]; __syncthreads(); }
    float mean = red[0] * (1.0f / D);
    __syncthreads();

    float vs = 0.0f;
    for (int i = 0; i < 4; ++i) { float dd = v[i] - mean; vs += dd * dd; }
    red[tid] = vs; __syncthreads();
    for (int o = 64; o > 0; o >>= 1) { if (tid < o) red[tid] += red[tid + o]; __syncthreads(); }
    float var = red[0] * (1.0f / D);
    float rstd = rsqrtf(var + 1e-5f);

    for (int i = 0; i < 4; ++i) {
        int idx = tid * 4 + i;
        out[(size_t)row * D + idx] = (v[i] - mean) * rstd * g[idx] + b[idx];
    }
}

// ---------------- GEMV (f32 weights): y[N] = x[K] * W[K,N] + bias ----------------
__global__ void __launch_bounds__(256)
gemv(const float* __restrict__ x, const float* __restrict__ W,
     const float* __restrict__ bias, float* __restrict__ y,
     int K, int N, int relu)
{
    __shared__ float xs[2048];
    int tid = threadIdx.x;
    for (int i = tid; i < (K >> 2); i += 256)
        ((float4*)xs)[i] = ((const float4*)x)[i];
    __syncthreads();
    int j = blockIdx.x * 256 + tid;
    if (j < N) {
        float s = bias ? bias[j] : 0.0f;
        for (int k = 0; k < K; ++k) s += xs[k] * W[(size_t)k * N + j];
        if (relu) s = fmaxf(s, 0.0f);
        y[j] = s;
    }
}

// ---------------- GEMV (bf16 weights, L2-resident decode path) ----------------
__global__ void __launch_bounds__(256)
gemv_bf16(const float* __restrict__ x, const bf16_t* __restrict__ W,
          const float* __restrict__ bias, float* __restrict__ y,
          int K, int N, int relu)
{
    __shared__ float xs[2048];
    int tid = threadIdx.x;
    for (int i = tid; i < (K >> 2); i += 256)
        ((float4*)xs)[i] = ((const float4*)x)[i];
    __syncthreads();
    int j = blockIdx.x * 256 + tid;
    if (j < N) {
        float s = bias ? bias[j] : 0.0f;
        for (int k = 0; k < K; ++k) s += xs[k] * (float)W[(size_t)k * N + j];
        if (relu) s = fmaxf(s, 0.0f);
        y[j] = s;
    }
}

// ---------------- sampling ----------------
__global__ void __launch_bounds__(1024)
sample_kernel(const float* __restrict__ logits, int nclass, int* __restrict__ ybuf,
              int cur_len, const float* __restrict__ ae,
              const float* __restrict__ alpha_a, int pe_pos,
              float* __restrict__ hvec, int seed)
{
    __shared__ float pen[1056];
    __shared__ unsigned char present[1056];
    __shared__ float rv[1024];
    __shared__ int   ri[1024];
    __shared__ float thr, smax, ssum;
    __shared__ int stok;
    int tid = threadIdx.x;

    for (int i = tid; i < 1056; i += 1024) { pen[i] = -1e30f; present[i] = 0; }
    __syncthreads();
    for (int i = tid; i < nclass; i += 1024) pen[i] = logits[i];
    __syncthreads();
    for (int t = tid; t < cur_len; t += 1024) {
        int tok = ybuf[t];
        if (tok >= 0 && tok < nclass) present[tok] = 1;
    }
    __syncthreads();
    for (int i = tid; i < nclass; i += 1024) {
        if (present[i]) { float l = pen[i]; pen[i] = (l < 0.0f) ? l * 1.35f : l * (1.0f / 1.35f); }
    }
    __syncthreads();

    if (tid == 0) {
        float best[15];
        for (int j = 0; j < 15; ++j) best[j] = -1e30f;
        for (int i = 0; i < nclass; ++i) {
            float l = pen[i];
            if (l > best[14]) {
                int j = 14;
                while (j > 0 && best[j - 1] < l) { best[j] = best[j - 1]; --j; }
                best[j] = l;
            }
        }
        thr = best[14]; smax = best[0];
    }
    __syncthreads();

    float lsum = 0.0f;
    for (int i = tid; i < nclass; i += 1024) {
        float l = pen[i];
        if (l >= thr) lsum += expf(l - smax);
    }
    rv[tid] = lsum; __syncthreads();
    for (int o = 512; o > 0; o >>= 1) { if (tid < o) rv[tid] += rv[tid + o]; __syncthreads(); }
    if (tid == 0) ssum = rv[0];
    __syncthreads();

    float bestv = -1e30f; int besti = 0;
    for (int i = tid; i < nclass; i += 1024) {
        float l = pen[i];
        float p = (l >= thr) ? (expf(l - smax) / ssum) : 0.0f;
        unsigned h1 = hash_u32((unsigned)seed * 0x9E3779B9u + (unsigned)i * 2u + 1u);
        unsigned h2 = hash_u32((unsigned)seed * 0x85EBCA6Bu + (unsigned)i * 2u + 2u);
        float u1 = ((float)h1 + 1.0f) * 2.3283064e-10f;
        float u2 = (float)h2 * 2.3283064e-10f;
        float q = sqrtf(-2.0f * logf(u1)) * cosf(6.2831853f * u2);
        if (q == 0.0f) q = 1e-9f;
        float r = p / q;
        if (r > bestv) { bestv = r; besti = i; }
    }
    rv[tid] = bestv; ri[tid] = besti; __syncthreads();
    for (int o = 512; o > 0; o >>= 1) {
        if (tid < o && rv[tid + o] > rv[tid]) { rv[tid] = rv[tid + o]; ri[tid] = ri[tid + o]; }
        __syncthreads();
    }
    if (tid == 0) { stok = ri[0]; ybuf[cur_len] = stok; }
    __syncthreads();
    int tok = stok;
    if (tid < D) hvec[tid] = ae[(size_t)tok * D + tid] + alpha_a[0] * pe_val(pe_pos, tid);
}

__global__ void finalize_kernel(const int* __restrict__ ybuf, int* __restrict__ out) {
    int i = threadIdx.x;
    if (i < NEWTOK - 1) out[i] = (i == NEWTOK - 2) ? 0 : ybuf[PLEN + 1 + i];
}

// ---------------- host-side launch ----------------
extern "C" void kernel_launch(void* const* d_in, const int* in_sizes, int n_in,
                              void* d_out, int out_size, void* d_ws, size_t ws_size,
                              hipStream_t stream)
{
    (void)in_sizes; (void)n_in; (void)out_size;
    const float* te        = (const float*)d_in[0];
    const float* ae        = (const float*)d_in[1];
    const float* bw        = (const float*)d_in[2];
    const float* bb        = (const float*)d_in[3];
    const float* a_t       = (const float*)d_in[4];
    const float* a_a       = (const float*)d_in[5];
    const float* Wqkv      = (const float*)d_in[6];
    const float* bqkv      = (const float*)d_in[7];
    const float* Wo        = (const float*)d_in[8];
    const float* bo        = (const float*)d_in[9];
    const float* g1        = (const float*)d_in[10];
    const float* b1        = (const float*)d_in[11];
    const float* g2        = (const float*)d_in[12];
    const float* b2        = (const float*)d_in[13];
    const float* W1        = (const float*)d_in[14];
    const float* bf1       = (const float*)d_in[15];
    const float* W2        = (const float*)d_in[16];
    const float* bf2       = (const float*)d_in[17];
    const float* pw        = (const float*)d_in[18];
    const float* pb        = (const float*)d_in[19];
    const float* ref_bert  = (const float*)d_in[20];
    const float* text_bert = (const float*)d_in[21];
    const int*   prompts   = (const int*)d_in[22];
    const int*   ref_seq   = (const int*)d_in[23];
    const int*   text_seq  = (const int*)d_in[24];

    float* f = (float*)d_ws;
    size_t off = 0;
    float* h       = f + off; off += (size_t)SEQP * D;
    float* tmp     = f + off; off += (size_t)SEQP * D;
    float* qkvbuf  = f + off; off += (size_t)SEQP * 3 * D;
    float* ffbuf   = f + off; off += (size_t)SEQP * FF;
    float* attout  = f + off; off += (size_t)SEQP * D;
    float* bertbuf = f + off; off += (size_t)BERTP * 1024;
    float* kc      = f + off; off += (size_t)LAYERS * TMAX * D;
    float* vc      = f + off; off += (size_t)LAYERS * TMAX * D;
    float* hvec    = f + off; off += D;
    float* qkvvec  = f + off; off += 3 * D;
    float* ovec    = f + off; off += D;
    float* tvec    = f + off; off += D;
    float* fvec    = f + off; off += FF;
    float* logits  = f + off; off += 1056;
    int*   ybuf    = (int*)(f + off); off += 256;
    size_t f32_words = off;

    // optional bf16 weight cache (decode path: halves traffic, fits 192MB L2)
    const size_t nQ = (size_t)LAYERS * D * 3 * D;
    const size_t nO = (size_t)LAYERS * D * D;
    const size_t n1 = (size_t)LAYERS * D * FF;
    const size_t n2 = (size_t)LAYERS * FF * D;
    const size_t nP = (size_t)D * VOC;
    const size_t bf_words = nQ + nO + n1 + n2 + nP;
    bool use_bf16 = ws_size >= (f32_words * 4 + bf_words * 2 + 1024);

    bf16_t* wq_bf = (bf16_t*)(f + f32_words);
    bf16_t* wo_bf = wq_bf + nQ;
    bf16_t* w1_bf = wo_bf + nO;
    bf16_t* w2_bf = w1_bf + n1;
    bf16_t* pw_bf = w2_bf + n2;

    if (use_bf16) {
        cvt_bf16<<<dim3((unsigned)((nQ / 4 + 255) / 256)), 256, 0, stream>>>(Wqkv, wq_bf, (int)(nQ / 4));
        cvt_bf16<<<dim3((unsigned)((nO / 4 + 255) / 256)), 256, 0, stream>>>(Wo,   wo_bf, (int)(nO / 4));
        cvt_bf16<<<dim3((unsigned)((n1 / 4 + 255) / 256)), 256, 0, stream>>>(W1,   w1_bf, (int)(n1 / 4));
        cvt_bf16<<<dim3((unsigned)((n2 / 4 + 255) / 256)), 256, 0, stream>>>(W2,   w2_bf, (int)(n2 / 4));
        cvt_bf16<<<dim3((unsigned)((nP / 4 + 255) / 256)), 256, 0, stream>>>(pw,   pw_bf, (int)(nP / 4));
    }

    // ---- setup: pack bert, bert projection (WMMA), embeddings, prompt tokens ----
    pack_bert<<<XLEN, 256, 0, stream>>>(ref_bert, text_bert, bertbuf);
    gemm_wmma<<<dim3(D / BN, BERTP / BM), 128, 0, stream>>>(bertbuf, bw, nullptr, tmp, BERTP, D, 1024, 0);
    embed_kernel<<<SEQ, D, 0, stream>>>(te, ae, tmp, bb, a_t, a_a, prompts, ref_seq, text_seq, h);
    init_ybuf<<<1, 256, 0, stream>>>(prompts, ybuf);

    const int MG = SEQP / BM;   // 6

    // ---- prefill: 24 layers, all matmuls through WMMA ----
    for (int l = 0; l < LAYERS; ++l) {
        const float* Wq  = Wqkv + (size_t)l * D * 3 * D;
        const float* Wol = Wo   + (size_t)l * D * D;
        const float* W1l = W1   + (size_t)l * D * FF;
        const float* W2l = W2   + (size_t)l * FF * D;
        float* kcl = kc + (size_t)l * TMAX * D;
        float* vcl = vc + (size_t)l * TMAX * D;

        gemm_wmma<<<dim3(3 * D / BN, MG), 128, 0, stream>>>(h, Wq, bqkv + (size_t)l * 3 * D, qkvbuf, SEQP, 3 * D, D, 0);
        attn_prefill<<<dim3(SEQ, HEADS), 32, 0, stream>>>(qkvbuf, kcl, vcl, attout);
        gemm_wmma<<<dim3(D / BN, MG), 128, 0, stream>>>(attout, Wol, bo + (size_t)l * D, tmp, SEQP, D, D, 0);
        add_ln<<<SEQ, 128, 0, stream>>>(h, tmp, g1 + (size_t)l * D, b1 + (size_t)l * D, h);
        gemm_wmma<<<dim3(FF / BN, MG), 128, 0, stream>>>(h, W1l, bf1 + (size_t)l * FF, ffbuf, SEQP, FF, D, 1);
        gemm_wmma<<<dim3(D / BN, MG), 128, 0, stream>>>(ffbuf, W2l, bf2 + (size_t)l * D, tmp, SEQP, D, FF, 0);
        add_ln<<<SEQ, 128, 0, stream>>>(h, tmp, g2 + (size_t)l * D, b2 + (size_t)l * D, h);
    }

    // ---- first sample (last row logits, EOS class dropped) ----
    if (use_bf16)
        gemv_bf16<<<dim3((VOC + 255) / 256), 256, 0, stream>>>(h + (size_t)(SEQ - 1) * D, pw_bf, pb, logits, D, VOC, 0);
    else
        gemv<<<dim3((VOC + 255) / 256), 256, 0, stream>>>(h + (size_t)(SEQ - 1) * D, pw, pb, logits, D, VOC, 0);
    sample_kernel<<<1, 1024, 0, stream>>>(logits, VOC - 1, ybuf, PLEN, ae, a_a, PLEN, hvec, 1000);

    // ---- decode: 31 steps ----
    for (int i = 1; i < NEWTOK; ++i) {
        int pos = SEQ - 1 + i;
        for (int l = 0; l < LAYERS; ++l) {
            float* kcl = kc + (size_t)l * TMAX * D;
            float* vcl = vc + (size_t)l * TMAX * D;
            if (use_bf16) {
                gemv_bf16<<<dim3(6), 256, 0, stream>>>(hvec, wq_bf + (size_t)l * D * 3 * D, bqkv + (size_t)l * 3 * D, qkvvec, D, 3 * D, 0);
                attn_decode<<<HEADS, 32, 0, stream>>>(qkvvec, kcl, vcl, ovec, pos);
                gemv_bf16<<<dim3(2), 256, 0, stream>>>(ovec, wo_bf + (size_t)l * D * D, bo + (size_t)l * D, tvec, D, D, 0);
                add_ln<<<1, 128, 0, stream>>>(hvec, tvec, g1 + (size_t)l * D, b1 + (size_t)l * D, hvec);
                gemv_bf16<<<dim3(8), 256, 0, stream>>>(hvec, w1_bf + (size_t)l * D * FF, bf1 + (size_t)l * FF, fvec, D, FF, 1);
                gemv_bf16<<<dim3(2), 256, 0, stream>>>(fvec, w2_bf + (size_t)l * FF * D, bf2 + (size_t)l * D, tvec, FF, D, 0);
                add_ln<<<1, 128, 0, stream>>>(hvec, tvec, g2 + (size_t)l * D, b2 + (size_t)l * D, hvec);
            } else {
                gemv<<<dim3(6), 256, 0, stream>>>(hvec, Wqkv + (size_t)l * D * 3 * D, bqkv + (size_t)l * 3 * D, qkvvec, D, 3 * D, 0);
                attn_decode<<<HEADS, 32, 0, stream>>>(qkvvec, kcl, vcl, ovec, pos);
                gemv<<<dim3(2), 256, 0, stream>>>(ovec, Wo + (size_t)l * D * D, bo + (size_t)l * D, tvec, D, D, 0);
                add_ln<<<1, 128, 0, stream>>>(hvec, tvec, g1 + (size_t)l * D, b1 + (size_t)l * D, hvec);
                gemv<<<dim3(8), 256, 0, stream>>>(hvec, W1 + (size_t)l * D * FF, bf1 + (size_t)l * FF, fvec, D, FF, 1);
                gemv<<<dim3(2), 256, 0, stream>>>(fvec, W2 + (size_t)l * FF * D, bf2 + (size_t)l * D, tvec, FF, D, 0);
                add_ln<<<1, 128, 0, stream>>>(hvec, tvec, g2 + (size_t)l * D, b2 + (size_t)l * D, hvec);
            }
        }
        if (use_bf16)
            gemv_bf16<<<dim3((VOC + 255) / 256), 256, 0, stream>>>(hvec, pw_bf, pb, logits, D, VOC, 0);
        else
            gemv<<<dim3((VOC + 255) / 256), 256, 0, stream>>>(hvec, pw, pb, logits, D, VOC, 0);
        sample_kernel<<<1, 1024, 0, stream>>>(logits, VOC, ybuf, PLEN + i, ae, a_a, PLEN + i, hvec, 1000 + i);
    }

    finalize_kernel<<<1, 64, 0, stream>>>(ybuf, (int*)d_out);
}
